// RNN_76510547410955
// MI455X (gfx1250) — compile-verified, matrix operand order checked by reading, not run on hardware
//
#include <hip/hip_runtime.h>
#include <hip/hip_bf16.h>

#define BS 512
#define T  100
#define ND 10
#define DS 6
#define IN 256
#define H  256
#define SA 2816        // IN*ND + H
#define G4 1024        // 4*H

typedef __bf16 bf16_t;
typedef __attribute__((ext_vector_type(8)))  __bf16 bf16x8;
typedef __attribute__((ext_vector_type(16))) __bf16 v16bf;
typedef __attribute__((ext_vector_type(8)))  float  v8f;

// Load one 16x32 bf16 WMMA fragment (A or B), row-major K-contiguous source.
// Lane l (l16 = l&15, half = l>>4) holds row (rowBase+l16),
// K = k + half*8 + {0..7}  and  k + 16 + half*8 + {0..7}.
__device__ __forceinline__ v16bf load_frag(const bf16_t* __restrict__ base,
                                           int ld, int row, int k0) {
  const bf16x8* p = reinterpret_cast<const bf16x8*>(base + (size_t)row * ld + k0);
  bf16x8 lo = p[0];   // K = k0 .. k0+7
  bf16x8 hi = p[2];   // K = k0+16 .. k0+23
  return __builtin_shufflevector(lo, hi, 0,1,2,3,4,5,6,7,8,9,10,11,12,13,14,15);
}

__device__ __forceinline__ v8f wmma_bf16(v16bf a, v16bf b, v8f c) {
  return __builtin_amdgcn_wmma_f32_16x16x32_bf16(false, a, false, b,
                                                 (short)0, c, false, false);
}

__device__ __forceinline__ float sigmoidf(float x) {
  return 1.0f / (1.0f + __expf(-x));
}

// C/D tile layout: vgpr i, lanes 0-15 -> (M=i, N=lane); lanes 16-31 -> (M=i+8, N=lane-16).
__device__ __forceinline__ void store_relu_bf16(v8f acc, const float* __restrict__ bias,
                                                bf16_t* __restrict__ y, int ld,
                                                int mBase, int half, int l16, int nBase) {
  int col = nBase + l16;
  float b = bias[col];
#pragma unroll
  for (int i = 0; i < 8; ++i) {
    float v = acc[i] + b;
    v = v > 0.0f ? v : 0.0f;
    y[(size_t)(mBase + half * 8 + i) * ld + col] = (bf16_t)v;
  }
}

// ---------------- prep: fp32 weights -> bf16 in workspace, zero state ----------------
__global__ __launch_bounds__(256) void prep_kernel(
    const float* __restrict__ W1, const float* __restrict__ W2,
    const float* __restrict__ Wv, const float* __restrict__ Wih,
    const float* __restrict__ Whh,
    bf16_t* __restrict__ W1b, bf16_t* __restrict__ W2b, bf16_t* __restrict__ Wvb,
    bf16_t* __restrict__ Wihb, bf16_t* __restrict__ Whhb,
    bf16_t* __restrict__ h0, bf16_t* __restrict__ inp0, float* __restrict__ c0) {
  const int NW1 = SA * SA;
  for (int i = blockIdx.x * blockDim.x + threadIdx.x; i < NW1;
       i += gridDim.x * blockDim.x) {
    W1b[i] = (bf16_t)W1[i];
    if (i < G4 * IN) { Wihb[i] = (bf16_t)Wih[i]; Whhb[i] = (bf16_t)Whh[i]; }
    if (i < IN * SA) W2b[i] = (bf16_t)W2[i];
    if (i < 16 * SA) {                       // Wv padded from ND=10 to 16 rows
      int r = i / SA;
      Wvb[i] = (r < ND) ? (bf16_t)Wv[i] : (bf16_t)0.0f;
    }
    if (i < BS * H) { h0[i] = (bf16_t)0.0f; inp0[i] = (bf16_t)0.0f; c0[i] = 0.0f; }
  }
}

// ---------------- LSTM: gates = inp@Wih^T + h@Whh^T + b ; update c,h ----------------
// grid (2,16), block 256 (8 waves). Wave: 32 rows x 16 cols of EACH gate
// (8 accumulators; A fragments shared across the 4 gate B fragments).
__global__ __launch_bounds__(256) void lstm_kernel(
    const bf16_t* __restrict__ inp, const bf16_t* __restrict__ h_in,
    const bf16_t* __restrict__ Wih, const bf16_t* __restrict__ Whh,
    const float* __restrict__ b_ih, const float* __restrict__ b_hh,
    float* __restrict__ c, bf16_t* __restrict__ h_out, bf16_t* __restrict__ xbuf) {
  const int wave = threadIdx.x >> 5;
  const int lane = threadIdx.x & 31;
  const int l16  = lane & 15;
  const int half = lane >> 4;
  const int mBase = (blockIdx.x * 8 + wave) * 32;
  const int jBase = blockIdx.y * 16;
  const int aRow = mBase + l16;
  const int bRow = jBase + l16;

  v8f i0 = {}, i1 = {}, f0 = {}, f1 = {}, g0 = {}, g1 = {}, o0 = {}, o1 = {};
#pragma unroll
  for (int k = 0; k < IN; k += 32) {
    int k0 = k + half * 8;
    v16bf a0 = load_frag(inp, IN, aRow,      k0);
    v16bf a1 = load_frag(inp, IN, aRow + 16, k0);
    v16bf bI = load_frag(Wih, IN, bRow,       k0);
    v16bf bF = load_frag(Wih, IN, bRow + 256, k0);
    v16bf bG = load_frag(Wih, IN, bRow + 512, k0);
    v16bf bO = load_frag(Wih, IN, bRow + 768, k0);
    i0 = wmma_bf16(a0, bI, i0); i1 = wmma_bf16(a1, bI, i1);
    f0 = wmma_bf16(a0, bF, f0); f1 = wmma_bf16(a1, bF, f1);
    g0 = wmma_bf16(a0, bG, g0); g1 = wmma_bf16(a1, bG, g1);
    o0 = wmma_bf16(a0, bO, o0); o1 = wmma_bf16(a1, bO, o1);
  }
#pragma unroll
  for (int k = 0; k < H; k += 32) {
    int k0 = k + half * 8;
    v16bf a0 = load_frag(h_in, H, aRow,      k0);
    v16bf a1 = load_frag(h_in, H, aRow + 16, k0);
    v16bf bI = load_frag(Whh, H, bRow,       k0);
    v16bf bF = load_frag(Whh, H, bRow + 256, k0);
    v16bf bG = load_frag(Whh, H, bRow + 512, k0);
    v16bf bO = load_frag(Whh, H, bRow + 768, k0);
    i0 = wmma_bf16(a0, bI, i0); i1 = wmma_bf16(a1, bI, i1);
    f0 = wmma_bf16(a0, bF, f0); f1 = wmma_bf16(a1, bF, f1);
    g0 = wmma_bf16(a0, bG, g0); g1 = wmma_bf16(a1, bG, g1);
    o0 = wmma_bf16(a0, bO, o0); o1 = wmma_bf16(a1, bO, o1);
  }

  const int col = jBase + l16;
  float bi = b_ih[col]       + b_hh[col];
  float bf = b_ih[col + 256] + b_hh[col + 256];
  float bg = b_ih[col + 512] + b_hh[col + 512];
  float bo = b_ih[col + 768] + b_hh[col + 768];

#pragma unroll
  for (int rg = 0; rg < 2; ++rg) {
    const v8f& aI = rg ? i1 : i0;
    const v8f& aF = rg ? f1 : f0;
    const v8f& aG = rg ? g1 : g0;
    const v8f& aO = rg ? o1 : o0;
    int mB = mBase + rg * 16;
#pragma unroll
    for (int i = 0; i < 8; ++i) {
      int row = mB + half * 8 + i;
      size_t idx = (size_t)row * H + col;
      float iv = sigmoidf(aI[i] + bi);
      float fv = sigmoidf(aF[i] + bf);
      float gv = tanhf(aG[i] + bg);
      float ov = sigmoidf(aO[i] + bo);
      float cn = fv * c[idx] + iv * gv;
      c[idx] = cn;
      float hn = ov * tanhf(cn);
      bf16_t hb = (bf16_t)hn;
      h_out[idx] = hb;                                   // next step's h input
      xbuf[(size_t)row * SA + (IN * ND) + col] = hb;     // concat slot for MLP
    }
  }
}

// ---------------- embed: emb = d @ Wd^T + bd -> xbuf[:, :2560] (K=6, scalar) ---------
__global__ __launch_bounds__(256) void embed_kernel(
    const float* __restrict__ desc, const float* __restrict__ Wd,
    const float* __restrict__ bd, bf16_t* __restrict__ xbuf, int t) {
  int idx = blockIdx.x * blockDim.x + threadIdx.x;
  if (idx >= BS * IN * ND) return;
  int b   = idx / (IN * ND);
  int rem = idx % (IN * ND);
  int nd  = rem / IN;
  int j   = rem % IN;
  const float* dp = desc + (((size_t)b * T + t) * ND + nd) * DS;
  const float* wp = Wd + (size_t)j * DS;
  float s = bd[j];
#pragma unroll
  for (int q = 0; q < DS; ++q) s += dp[q] * wp[q];
  xbuf[(size_t)b * SA + rem] = (bf16_t)s;
}

// ---------------- mlp1: x1 = relu(x @ W1^T + b1), [512,2816]x[2816,2816] -------------
// grid (2,44), block 256 (8 waves). Wave: 32 rows x 64 cols
// (2 A frags x 4 B frags -> 8 WMMAs per k-tile; 1.5 b128 loads per WMMA).
__global__ __launch_bounds__(256) void mlp1_kernel(
    const bf16_t* __restrict__ x, const bf16_t* __restrict__ W1,
    const float* __restrict__ b1, bf16_t* __restrict__ y) {
  const int wave = threadIdx.x >> 5;
  const int lane = threadIdx.x & 31;
  const int l16  = lane & 15;
  const int half = lane >> 4;
  const int mBase = (blockIdx.x * 8 + wave) * 32;
  const int nBase = blockIdx.y * 64;
  const int aRow = mBase + l16;
  const int bRow = nBase + l16;

  v8f c00 = {}, c01 = {}, c02 = {}, c03 = {};
  v8f c10 = {}, c11 = {}, c12 = {}, c13 = {};
  for (int k = 0; k < SA; k += 32) {
    int k0 = k + half * 8;
    v16bf a0 = load_frag(x, SA, aRow,      k0);
    v16bf a1 = load_frag(x, SA, aRow + 16, k0);
    v16bf b0 = load_frag(W1, SA, bRow,      k0);
    v16bf b1f = load_frag(W1, SA, bRow + 16, k0);
    v16bf b2 = load_frag(W1, SA, bRow + 32, k0);
    v16bf b3 = load_frag(W1, SA, bRow + 48, k0);
    c00 = wmma_bf16(a0, b0, c00);  c10 = wmma_bf16(a1, b0, c10);
    c01 = wmma_bf16(a0, b1f, c01); c11 = wmma_bf16(a1, b1f, c11);
    c02 = wmma_bf16(a0, b2, c02);  c12 = wmma_bf16(a1, b2, c12);
    c03 = wmma_bf16(a0, b3, c03);  c13 = wmma_bf16(a1, b3, c13);
  }
  store_relu_bf16(c00, b1, y, SA, mBase,      half, l16, nBase);
  store_relu_bf16(c01, b1, y, SA, mBase,      half, l16, nBase + 16);
  store_relu_bf16(c02, b1, y, SA, mBase,      half, l16, nBase + 32);
  store_relu_bf16(c03, b1, y, SA, mBase,      half, l16, nBase + 48);
  store_relu_bf16(c10, b1, y, SA, mBase + 16, half, l16, nBase);
  store_relu_bf16(c11, b1, y, SA, mBase + 16, half, l16, nBase + 16);
  store_relu_bf16(c12, b1, y, SA, mBase + 16, half, l16, nBase + 32);
  store_relu_bf16(c13, b1, y, SA, mBase + 16, half, l16, nBase + 48);
}

// ---------------- mlp2: nxt = relu(x1@W2^T+b2) -> inp ; a = x1@Wv^T+bv -> out --------
// grid (2,17): nTile 0..15 -> W2, nTile 16 -> padded Wv. Wave: 32 rows x 16 cols.
__global__ __launch_bounds__(256) void mlp2_kernel(
    const bf16_t* __restrict__ x1, const bf16_t* __restrict__ W2,
    const bf16_t* __restrict__ Wv, const float* __restrict__ b2,
    const float* __restrict__ bv, bf16_t* __restrict__ inp_out,
    float* __restrict__ out, int t) {
  const int wave = threadIdx.x >> 5;
  const int lane = threadIdx.x & 31;
  const int l16  = lane & 15;
  const int half = lane >> 4;
  const int mBase = (blockIdx.x * 8 + wave) * 32;
  const int nTile = blockIdx.y;

  const bf16_t* B = (nTile < 16) ? W2 : Wv;
  const int bRow  = (nTile < 16) ? (nTile * 16 + l16) : l16;
  const int aRow  = mBase + l16;

  v8f acc0 = {}, acc1 = {};
  for (int k = 0; k < SA; k += 32) {
    int k0 = k + half * 8;
    v16bf bf = load_frag(B, SA, bRow, k0);
    acc0 = wmma_bf16(load_frag(x1, SA, aRow,      k0), bf, acc0);
    acc1 = wmma_bf16(load_frag(x1, SA, aRow + 16, k0), bf, acc1);
  }

  if (nTile < 16) {
    store_relu_bf16(acc0, b2, inp_out, IN, mBase,      half, l16, nTile * 16);
    store_relu_bf16(acc1, b2, inp_out, IN, mBase + 16, half, l16, nTile * 16);
  } else {
    int n = l16;
    if (n < ND) {
      float bias = bv[n];
#pragma unroll
      for (int rg = 0; rg < 2; ++rg) {
        const v8f& acc = rg ? acc1 : acc0;
        int mB = mBase + rg * 16;
#pragma unroll
        for (int i = 0; i < 8; ++i) {
          int row = mB + half * 8 + i;
          out[(size_t)row * T * ND + (size_t)t * ND + n] = acc[i] + bias;  // no sigmoid (training mode)
        }
      }
    }
  }
}

extern "C" void kernel_launch(void* const* d_in, const int* in_sizes, int n_in,
                              void* d_out, int out_size, void* d_ws, size_t ws_size,
                              hipStream_t stream) {
  (void)in_sizes; (void)n_in; (void)out_size; (void)ws_size;
  const float* desc = (const float*)d_in[0];
  const float* Wd   = (const float*)d_in[1];
  const float* bd   = (const float*)d_in[2];
  const float* W1   = (const float*)d_in[3];
  const float* b1   = (const float*)d_in[4];
  const float* W2   = (const float*)d_in[5];
  const float* b2   = (const float*)d_in[6];
  const float* Wv   = (const float*)d_in[7];
  const float* bv   = (const float*)d_in[8];
  const float* Wih  = (const float*)d_in[9];
  const float* Whh  = (const float*)d_in[10];
  const float* b_ih = (const float*)d_in[11];
  const float* b_hh = (const float*)d_in[12];
  float* out = (float*)d_out;

  char* ws = (char*)d_ws;
  size_t off = 0;
  auto alloc = [&](size_t bytes) -> char* {
    char* p = ws + off;
    off = (off + bytes + 255) & ~(size_t)255;
    return p;
  };
  bf16_t* W1b   = (bf16_t*)alloc((size_t)SA * SA * 2);   // 15.9 MB
  bf16_t* W2b   = (bf16_t*)alloc((size_t)IN * SA * 2);
  bf16_t* Wvb   = (bf16_t*)alloc((size_t)16 * SA * 2);
  bf16_t* Wihb  = (bf16_t*)alloc((size_t)G4 * IN * 2);
  bf16_t* Whhb  = (bf16_t*)alloc((size_t)G4 * H * 2);
  bf16_t* xbuf  = (bf16_t*)alloc((size_t)BS * SA * 2);
  bf16_t* x1buf = (bf16_t*)alloc((size_t)BS * SA * 2);
  bf16_t* inpb  = (bf16_t*)alloc((size_t)BS * IN * 2);
  bf16_t* h_a   = (bf16_t*)alloc((size_t)BS * H * 2);
  bf16_t* h_b   = (bf16_t*)alloc((size_t)BS * H * 2);
  float*  cbuf  = (float*) alloc((size_t)BS * H * 4);

  prep_kernel<<<1024, 256, 0, stream>>>(W1, W2, Wv, Wih, Whh,
                                        W1b, W2b, Wvb, Wihb, Whhb,
                                        h_a, inpb, cbuf);

  for (int t = 0; t < T; ++t) {
    const bf16_t* h_in = (t & 1) ? h_b : h_a;
    bf16_t* h_out      = (t & 1) ? h_a : h_b;
    lstm_kernel<<<dim3(2, 16), 256, 0, stream>>>(inpb, h_in, Wihb, Whhb,
                                                 b_ih, b_hh, cbuf, h_out, xbuf);
    embed_kernel<<<(BS * IN * ND + 255) / 256, 256, 0, stream>>>(desc, Wd, bd, xbuf, t);
    mlp1_kernel<<<dim3(2, 44), 256, 0, stream>>>(xbuf, W1b, b1, x1buf);
    mlp2_kernel<<<dim3(2, 17), 256, 0, stream>>>(x1buf, W2b, Wvb, b2, bv, inpb, out, t);
  }
}
